// CausalSelfAttention_32280974197343
// MI455X (gfx1250) — matrix operand order repaired and checked by reading, not compiled, not run on hardware
//
#include <hip/hip_runtime.h>
#include <hip/hip_bf16.h>
#include <cstdint>

typedef __attribute__((ext_vector_type(16))) __bf16 v16bf;
typedef __attribute__((ext_vector_type(8)))  float  v8f;

#define N_EMBD 1024
#define NHEAD  16
#define HDIM   64
#define BBATCH 4
#define TSEQ   2048

static __device__ __forceinline__ __bf16 f2bf(float f) {
  union { float f; unsigned u; } a; a.f = f;
  unsigned u = a.u + (((a.u >> 16) & 1u) + 0x7FFFu);   // round-to-nearest-even
  union { unsigned short s; __bf16 b; } c; c.s = (unsigned short)(u >> 16);
  return c.b;
}

union FragU { v16bf v; uint4 u[2]; };

// A-matrix fragment (16x32 bf16): lane m = lane&15, half h = lane>>4.
// Elements 0..7  <- K = h*8 + 0..7 ; elements 8..15 <- K = 16 + h*8 + 0..7.
static __device__ __forceinline__ v16bf load_fragA(const __bf16* row, int h) {
  FragU f;
  f.u[0] = *reinterpret_cast<const uint4*>(row + h * 8);
  f.u[1] = *reinterpret_cast<const uint4*>(row + 16 + h * 8);
  return f.v;
}
// B-matrix fragment (32x16 bf16), B stored N-major ([n][k]):
// lane n = lane&15, half h = lane>>4; elements 0..7 <- K = h*8 + 0..7 ; elements 8..15 <- K = 16 + h*8 + 0..7.
static __device__ __forceinline__ v16bf load_fragB(const __bf16* row, int h) {
  FragU f;
  f.u[0] = *reinterpret_cast<const uint4*>(row + h * 8);
  f.u[1] = *reinterpret_cast<const uint4*>(row + 16 + h * 8);
  return f.v;
}

// CDNA5 async HBM->LDS copy (16B per lane), tracked by ASYNCcnt.
// Generic pointers to __shared__ hold the LDS byte offset in their low 32 bits
// (flat LDS aperture; the low 32 bits give the LDS address).
static __device__ __forceinline__ void async_b128(void* lds_dst, const void* gsrc) {
  unsigned lds = (unsigned)(size_t)lds_dst;
  unsigned long long ga = (unsigned long long)(size_t)gsrc;
  asm volatile("global_load_async_to_lds_b128 %0, %1, off"
               :: "v"(lds), "v"(ga) : "memory");
}
static __device__ __forceinline__ void wait_async0() {
  asm volatile("s_wait_asynccnt 0x0" ::: "memory");
}

// ---------------- precision conversion kernels ----------------
__global__ void csa_conv_bf16(const float* __restrict__ src,
                              __bf16* __restrict__ dst, size_t n) {
  size_t i = (size_t)blockIdx.x * blockDim.x + threadIdx.x;
  size_t stride = (size_t)gridDim.x * blockDim.x;
  for (; i < n; i += stride) dst[i] = f2bf(src[i]);
}

__global__ void csa_transpose_bf16(const float* __restrict__ W,
                                   __bf16* __restrict__ Wt, int K, int N) {
  size_t i = (size_t)blockIdx.x * blockDim.x + threadIdx.x;
  size_t tot = (size_t)K * N;
  size_t stride = (size_t)gridDim.x * blockDim.x;
  for (; i < tot; i += stride) {
    size_t kk = i / N, nn = i % N;
    Wt[nn * (size_t)K + kk] = f2bf(W[i]);
  }
}

// ---------------- tiled bf16 GEMM: C = A[MxK] * Bt[NxK]^T + bias ----------------
// MODE 0: write fp32 C.   MODE 1: scatter into q/k [B,H,T,D] and v^T [B,H,D,T].
// 128x128 tile, BK=32, async double-buffered LDS, one barrier per K-step.
template <int MODE>
__global__ __launch_bounds__(256) void csa_gemm_bf16(
    const __bf16* __restrict__ A, const __bf16* __restrict__ Bt,
    const float* __restrict__ bias, int M, int N, int K,
    float* __restrict__ outF, __bf16* __restrict__ qOut,
    __bf16* __restrict__ kOut, __bf16* __restrict__ vtOut) {
  __shared__ __bf16 As[2][128][32];
  __shared__ __bf16 Bs[2][128][32];   // n-major
  const int tid  = threadIdx.x;
  const int wave = tid >> 5, lane = tid & 31;
  const int h = lane >> 4, c = lane & 15;
  const int wr = wave & 3, wc = wave >> 2;      // 4x2 wave grid -> 32x64 per wave
  const int m0 = blockIdx.y * 128, n0 = blockIdx.x * 128;

  // staging coordinates for this thread (2x 16B chunks per matrix)
  const int row0 = tid >> 2,            off0 = (tid & 3) * 8;
  const int row1 = (tid + 256) >> 2,    off1 = ((tid + 256) & 3) * 8;

  const v8f vz = {0.f, 0.f, 0.f, 0.f, 0.f, 0.f, 0.f, 0.f};
  v8f acc[2][4];
  for (int mt = 0; mt < 2; ++mt)
    for (int nt = 0; nt < 4; ++nt) acc[mt][nt] = vz;

  const int nK = K >> 5;
  // prefetch tile 0 into buffer 0
  {
    async_b128(&As[0][row0][off0], A  + (size_t)(m0 + row0) * K + off0);
    async_b128(&Bs[0][row0][off0], Bt + (size_t)(n0 + row0) * K + off0);
    async_b128(&As[0][row1][off1], A  + (size_t)(m0 + row1) * K + off1);
    async_b128(&Bs[0][row1][off1], Bt + (size_t)(n0 + row1) * K + off1);
  }
  for (int ki = 0; ki < nK; ++ki) {
    wait_async0();
    __syncthreads();                      // tile[cur] visible; prior reads done
    const int cur = ki & 1;
    if (ki + 1 < nK) {                    // async-prefetch next tile
      const int nxt = cur ^ 1;
      const int kk = (ki + 1) << 5;
      async_b128(&As[nxt][row0][off0], A  + (size_t)(m0 + row0) * K + kk + off0);
      async_b128(&Bs[nxt][row0][off0], Bt + (size_t)(n0 + row0) * K + kk + off0);
      async_b128(&As[nxt][row1][off1], A  + (size_t)(m0 + row1) * K + kk + off1);
      async_b128(&Bs[nxt][row1][off1], Bt + (size_t)(n0 + row1) * K + kk + off1);
    }
    v16bf aF[2], bF[4];
    for (int mt = 0; mt < 2; ++mt) aF[mt] = load_fragA(&As[cur][wr * 32 + mt * 16 + c][0], h);
    for (int nt = 0; nt < 4; ++nt) bF[nt] = load_fragB(&Bs[cur][wc * 64 + nt * 16 + c][0], h);
    for (int mt = 0; mt < 2; ++mt)
      for (int nt = 0; nt < 4; ++nt)
        acc[mt][nt] = __builtin_amdgcn_wmma_f32_16x16x32_bf16(
            false, aF[mt], false, bF[nt], (short)0, acc[mt][nt], false, false);
  }

  for (int mt = 0; mt < 2; ++mt)
    for (int nt = 0; nt < 4; ++nt) {
      int colg = n0 + wc * 64 + nt * 16 + c;
      float bi = bias[colg];
      for (int i = 0; i < 8; ++i) {
        int rowg = m0 + wr * 32 + mt * 16 + i + h * 8;
        float val = acc[mt][nt][i] + bi;
        if (MODE == 0) {
          outF[(size_t)rowg * N + colg] = val;
        } else {
          int which = colg >> 10;            // 0:q 1:k 2:v   (N = 3072)
          int cc = colg & 1023;
          int head = cc >> 6, d = cc & 63;
          int b = rowg >> 11, t = rowg & 2047;
          size_t bh = (size_t)(b * NHEAD + head);
          __bf16 bv = f2bf(val);
          if (which == 0)      qOut[(bh * TSEQ + t) * HDIM + d] = bv;
          else if (which == 1) kOut[(bh * TSEQ + t) * HDIM + d] = bv;
          else                 vtOut[(bh * HDIM + d) * TSEQ + t] = bv;   // V^T
        }
      }
    }
}

// ---------------- flash-style causal attention ----------------
// grid: (B*H, T/128); 8 waves, each wave owns 16 queries x full D=64 and loops
// only to its own causal extent (no cross-wave sharing -> no block barriers).
__global__ __launch_bounds__(256) void csa_attn(
    const __bf16* __restrict__ q, const __bf16* __restrict__ k,
    const __bf16* __restrict__ vt, const int* __restrict__ amask,
    __bf16* __restrict__ y) {
  __shared__ __bf16 Ps[8][16][32];             // per-wave P re-layout buffer
  const int tid = threadIdx.x;
  const int wave = tid >> 5, lane = tid & 31;
  const int h = lane >> 4, c = lane & 15;
  const int bh = blockIdx.x;                   // 0..63
  const int qb = blockIdx.y;                   // 0..15
  const int b = bh >> 4, head = bh & 15;
  const int qbase = qb * 128 + wave * 16;
  const float NEG = -1e30f;

  const __bf16* Qp = q + (size_t)bh * TSEQ * HDIM;   // [t][d]
  const __bf16* Kp = k + (size_t)bh * TSEQ * HDIM;   // [t][d]
  const __bf16* Vp = vt + (size_t)bh * HDIM * TSEQ;  // [d][t]

  v16bf aQ[2];
  for (int ks = 0; ks < 2; ++ks)
    aQ[ks] = load_fragA(Qp + (size_t)(qbase + c) * HDIM + ks * 32, h);

  const v8f vz = {0.f, 0.f, 0.f, 0.f, 0.f, 0.f, 0.f, 0.f};
  v8f o[4] = {vz, vz, vz, vz};
  float m_run[8], l_run[8];
  for (int i = 0; i < 8; ++i) { m_run[i] = NEG; l_run[i] = 0.f; }

  const int nkb = (qbase + 16 + 31) >> 5;      // this wave's causal key blocks
  for (int kb = 0; kb < nkb; ++kb) {
    const int kbase = kb * 32;
    // L2 prefetch of next key block's K and V rows
    if (kb + 1 < nkb) {
      __builtin_prefetch(Kp + (size_t)(kbase + 32 + lane) * HDIM, 0, 0);
      __builtin_prefetch(Vp + (size_t)(lane & 63) * TSEQ + kbase + 32, 0, 0);
    }
    // ---- S = Q K^T (16x32 tile, fp32 accum) ----
    v8f s[2] = {vz, vz};
    for (int ks = 0; ks < 2; ++ks)
      for (int nt = 0; nt < 2; ++nt) {
        v16bf bK = load_fragB(Kp + (size_t)(kbase + nt * 16 + c) * HDIM + ks * 32, h);
        s[nt] = __builtin_amdgcn_wmma_f32_16x16x32_bf16(
            false, aQ[ks], false, bK, (short)0, s[nt], false, false);
      }
    // ---- mask + online softmax ----
    int colg[2], pm[2];
    for (int nt = 0; nt < 2; ++nt) {
      colg[nt] = kbase + nt * 16 + c;
      pm[nt] = amask[b * TSEQ + colg[nt]];
    }
    float p[2][8];
    for (int i = 0; i < 8; ++i) {
      int rowg = qbase + i + h * 8;
      float vmax = NEG;
      for (int nt = 0; nt < 2; ++nt) {
        float v = s[nt][i] * 0.125f;                 // 1/sqrt(64)
        if (colg[nt] > rowg || pm[nt] == 0) v = NEG;
        p[nt][i] = v;
        vmax = fmaxf(vmax, v);
      }
      for (int off = 1; off < 16; off <<= 1)
        vmax = fmaxf(vmax, __shfl_xor(vmax, off, 32));
      float mnew = fmaxf(m_run[i], vmax);
      float alpha = __expf(m_run[i] - mnew);
      float rsum = 0.f;
      for (int nt = 0; nt < 2; ++nt) {
        float e = __expf(p[nt][i] - mnew);
        p[nt][i] = e;
        rsum += e;
      }
      for (int off = 1; off < 16; off <<= 1)
        rsum += __shfl_xor(rsum, off, 32);
      l_run[i] = l_run[i] * alpha + rsum;
      m_run[i] = mnew;
      for (int nv = 0; nv < 4; ++nv) o[nv][i] *= alpha;
    }
    // ---- C-frag -> A-frag re-layout of P through per-wave LDS tile ----
    // In-wave DS ops are in-order (DScnt); wave_barrier only fences the compiler.
    for (int nt = 0; nt < 2; ++nt)
      for (int i = 0; i < 8; ++i)
        Ps[wave][i + h * 8][nt * 16 + c] = f2bf(p[nt][i]);
    __builtin_amdgcn_wave_barrier();
    v16bf aP = load_fragA(&Ps[wave][c][0], h);
    __builtin_amdgcn_wave_barrier();
    // ---- O += P V  (B = V^T rows, contiguous in t) ----
    for (int nv = 0; nv < 4; ++nv) {
      v16bf bV = load_fragB(Vp + (size_t)(nv * 16 + c) * TSEQ + kbase, h);
      o[nv] = __builtin_amdgcn_wmma_f32_16x16x32_bf16(
          false, aP, false, bV, (short)0, o[nv], false, false);
    }
  }
  // ---- normalize + write y as bf16 [B,T,C] ----
  for (int i = 0; i < 8; ++i) {
    float li = l_run[i];
    float inv = (li > 0.f) ? 1.0f / li : 0.f;
    int t = qbase + i + h * 8;
    for (int nv = 0; nv < 4; ++nv)
      y[((size_t)(b * TSEQ + t)) * N_EMBD + head * HDIM + nv * 16 + c] =
          f2bf(o[nv][i] * inv);
  }
}

extern "C" void kernel_launch(void* const* d_in, const int* in_sizes, int n_in,
                              void* d_out, int out_size, void* d_ws, size_t ws_size,
                              hipStream_t stream) {
  const float* x      = (const float*)d_in[0];
  const float* W_attn = (const float*)d_in[1];
  const float* b_attn = (const float*)d_in[2];
  const float* W_proj = (const float*)d_in[3];
  const float* b_proj = (const float*)d_in[4];
  const int*   amask  = (const int*)d_in[5];
  float* out = (float*)d_out;

  const size_t M = (size_t)BBATCH * TSEQ;            // 8192 rows
  unsigned char* p = (unsigned char*)d_ws;
  __bf16* x_bf = (__bf16*)p; p += M * N_EMBD * 2;                     // 16 MB
  __bf16* wat  = (__bf16*)p; p += (size_t)3 * N_EMBD * N_EMBD * 2;    //  6 MB  [3C][C]
  __bf16* wpt  = (__bf16*)p; p += (size_t)N_EMBD * N_EMBD * 2;        //  2 MB  [C][C]
  __bf16* qbf  = (__bf16*)p; p += M * N_EMBD * 2;                     // 16 MB  [B,H,T,D]
  __bf16* kbf  = (__bf16*)p; p += M * N_EMBD * 2;                     // 16 MB  [B,H,T,D]
  __bf16* vtbf = (__bf16*)p; p += M * N_EMBD * 2;                     // 16 MB  [B,H,D,T]
  __bf16* ybf  = (__bf16*)p; p += M * N_EMBD * 2;                     // 16 MB  [B,T,C]

  csa_conv_bf16<<<2048, 256, 0, stream>>>(x, x_bf, M * N_EMBD);
  csa_transpose_bf16<<<2048, 256, 0, stream>>>(W_attn, wat, N_EMBD, 3 * N_EMBD);
  csa_transpose_bf16<<<1024, 256, 0, stream>>>(W_proj, wpt, N_EMBD, N_EMBD);

  // qkv = x @ W_attn + b_attn  -> scatter q/k/v^T (bf16)
  csa_gemm_bf16<1><<<dim3(24, 64), 256, 0, stream>>>(
      x_bf, wat, b_attn, (int)M, 3 * N_EMBD, N_EMBD, nullptr, qbf, kbf, vtbf);

  // flash attention per (b,h) x 128-query blocks
  csa_attn<<<dim3(BBATCH * NHEAD, TSEQ / 128), 256, 0, stream>>>(
      qbf, kbf, vtbf, amask, ybf);

  // out = y @ W_proj + b_proj  (fp32 output)
  csa_gemm_bf16<0><<<dim3(8, 64), 256, 0, stream>>>(
      ybf, wpt, b_proj, (int)M, N_EMBD, N_EMBD, out, nullptr, nullptr, nullptr);
}
